// LSTMmodel_10917806867164
// MI455X (gfx1250) — matrix operand order repaired and checked by reading, not compiled, not run on hardware
//
#include <hip/hip_runtime.h>
#include <math.h>

// ---------------------------------------------------------------------------
// MI455X (gfx1250) 2-layer LSTM.
// Recurrent GEMMs use v_wmma_f32_16x16x32_bf16 (wave32 WMMA).
// Per time step: gates(64x4096) = x_t @ W_ih^T + h @ W_hh^T + b  (K=512/1024 + 1024)
// 64 blocks x 4 waves; wave g computes gate g for all 64 rows (disjoint weight
// rows -> minimal redundant L2 traffic; weights are L2-resident: 192MB L2).
// Gate exchange via LDS, LSTM cell update in-register.
// ---------------------------------------------------------------------------

typedef __attribute__((ext_vector_type(16))) __bf16 v16bf;
typedef __attribute__((ext_vector_type(8)))  float  v8f;

#define B_  64
#define T_  512
#define D_  512
#define H_  1024
#define G4_ 4096
#define FC_ 128
#define OUT_ 16

union AB16 { v16bf v; uint4 q[2]; };

__device__ __forceinline__ __bf16 f2bf(float f) {
  unsigned u = __builtin_bit_cast(unsigned, f);
  u += 0x7FFFu + ((u >> 16) & 1u);               // round-to-nearest-even
  unsigned short h = (unsigned short)(u >> 16);
  return __builtin_bit_cast(__bf16, h);
}
__device__ __forceinline__ float bf2f(__bf16 b) {
  unsigned u = ((unsigned)__builtin_bit_cast(unsigned short, b)) << 16;
  return __builtin_bit_cast(float, u);
}
__device__ __forceinline__ float sigm(float x) { return 1.0f / (1.0f + __expf(-x)); }

// ---------------- conversion / init kernels ----------------

__global__ __launch_bounds__(256) void k_f32_to_bf16(const float* __restrict__ in,
                                                     __bf16* __restrict__ out, size_t n) {
  size_t i = (size_t)blockIdx.x * blockDim.x + threadIdx.x;
  size_t stride = (size_t)gridDim.x * blockDim.x;
  for (; i < n; i += stride) out[i] = f2bf(in[i]);
}

__global__ __launch_bounds__(256) void k_bias_sum(const float* __restrict__ a,
                                                  const float* __restrict__ b,
                                                  float* __restrict__ out, int n) {
  int i = blockIdx.x * blockDim.x + threadIdx.x;
  if (i < n) out[i] = a[i] + b[i];
}

__global__ __launch_bounds__(256) void k_zero_state(__bf16* __restrict__ h,
                                                    float* __restrict__ c, int n) {
  int i = blockIdx.x * blockDim.x + threadIdx.x;
  if (i < n) { h[i] = f2bf(0.0f); c[i] = 0.0f; }
}

// ---------------- LSTM step kernel (WMMA bf16) ----------------
// X:   (B*T, KX) bf16, row index = b*T_ + t
// Wih: (4096, KX) bf16 row-major ; Whh: (4096, 1024) bf16 row-major
// bsum: b_ih + b_hh (4096) f32
// hprev/cprev -> hnext/cnext (64x1024); hs: (B*T,1024) sequence output (bf16)
template <int KX>
__global__ __launch_bounds__(128)
void k_lstm_step(const __bf16* __restrict__ X,
                 const __bf16* __restrict__ Wih,
                 const __bf16* __restrict__ Whh,
                 const float*  __restrict__ bsum,
                 const __bf16* __restrict__ hprev,
                 const float*  __restrict__ cprev,
                 __bf16* __restrict__ hnext,
                 float*  __restrict__ cnext,
                 __bf16* __restrict__ hs,
                 int t) {
  __shared__ float gl[4][64][16];                // 16 KB gate exchange

  const int lane = threadIdx.x & 31;
  const int g    = threadIdx.x >> 5;             // wave id == gate index (i,f,g,o)
  const int nl   = lane & 15;
  const int n    = (blockIdx.x << 4) + nl;       // hidden column
  const int khA  = (lane & 16) ? 8  : 0;         // A-fragment K sub-offset
  const int koB  = (lane & 16) ? 16 : 0;         // B-fragment K offset
  const int mhi  = (lane & 16) ? 8  : 0;         // C-fragment M offset

  const v8f zeroV = {0.f,0.f,0.f,0.f,0.f,0.f,0.f,0.f};
  v8f acc[4];
  #pragma unroll
  for (int mt = 0; mt < 4; ++mt) acc[mt] = zeroV;

  const __bf16* wih_row = Wih + (size_t)(g * H_ + n) * KX;
  const __bf16* whh_row = Whh + (size_t)(g * H_ + n) * H_;

  // ---- x_t @ W_ih^T ----
  for (int k0 = 0; k0 < KX; k0 += 32) {
    AB16 bf;
    bf.q[0] = *(const uint4*)(wih_row + k0 + khA);
    bf.q[1] = *(const uint4*)(wih_row + k0 + 16 + khA);
    #pragma unroll
    for (int mt = 0; mt < 4; ++mt) {
      const int m = mt * 16 + nl;                // m == batch index
      const __bf16* arow = X + ((size_t)m * T_ + t) * KX;
      AB16 af;
      af.q[0] = *(const uint4*)(arow + k0 + khA);
      af.q[1] = *(const uint4*)(arow + k0 + 16 + khA);
      acc[mt] = __builtin_amdgcn_wmma_f32_16x16x32_bf16(
          false, af.v, false, bf.v, (short)0, acc[mt], false, false);
    }
  }

  // ---- h_prev @ W_hh^T ----
  for (int k0 = 0; k0 < H_; k0 += 32) {
    AB16 bf;
    bf.q[0] = *(const uint4*)(whh_row + k0 + khA);
    bf.q[1] = *(const uint4*)(whh_row + k0 + 16 + khA);
    #pragma unroll
    for (int mt = 0; mt < 4; ++mt) {
      const __bf16* arow = hprev + (size_t)(mt * 16 + nl) * H_;
      AB16 af;
      af.q[0] = *(const uint4*)(arow + k0 + khA);
      af.q[1] = *(const uint4*)(arow + k0 + 16 + khA);
      acc[mt] = __builtin_amdgcn_wmma_f32_16x16x32_bf16(
          false, af.v, false, bf.v, (short)0, acc[mt], false, false);
    }
  }

  // ---- write gate tile (+bias) to LDS ----
  const float bs = bsum[g * H_ + n];
  #pragma unroll
  for (int mt = 0; mt < 4; ++mt) {
    #pragma unroll
    for (int r = 0; r < 8; ++r) {
      gl[g][mt * 16 + r + mhi][nl] = acc[mt][r] + bs;   // C layout: M = r + mhi
    }
  }
  __syncthreads();

  // ---- LSTM cell update: wave g handles rows [16g, 16g+16) ----
  {
    const int m0 = g * 16;
    #pragma unroll
    for (int r = 0; r < 8; ++r) {
      const int m = m0 + r + mhi;
      const float iv = gl[0][m][nl];
      const float fv = gl[1][m][nl];
      const float gv = gl[2][m][nl];
      const float ov = gl[3][m][nl];
      const float cp = cprev[(size_t)m * H_ + n];
      const float cn = sigm(fv) * cp + sigm(iv) * tanhf(gv);
      const float hn = sigm(ov) * tanhf(cn);
      cnext[(size_t)m * H_ + n] = cn;
      const __bf16 hb = f2bf(hn);
      hnext[(size_t)m * H_ + n] = hb;
      hs[((size_t)m * T_ + t) * H_ + n] = hb;
    }
  }
}

// ---------------- FC head ----------------

__global__ __launch_bounds__(128)
void k_fc1(const __bf16* __restrict__ hs1, const float* __restrict__ W,
           const float* __restrict__ b, float* __restrict__ out) {
  const int m = blockIdx.x;        // batch row
  const int nthr = threadIdx.x;    // 0..127 output feature
  const __bf16* row = hs1 + ((size_t)m * T_ + (T_ - 1)) * H_;
  const float* w = W + (size_t)nthr * H_;
  float s = 0.0f;
  for (int k = 0; k < H_; ++k) s += bf2f(row[k]) * w[k];
  s += b[nthr];
  out[m * FC_ + nthr] = s > 0.0f ? s : 0.0f;
}

__global__ __launch_bounds__(32)
void k_fc2_logsoftmax(const float* __restrict__ in, const float* __restrict__ W,
                      const float* __restrict__ b, float* __restrict__ out) {
  __shared__ float o[OUT_];
  __shared__ float lse;
  const int m = blockIdx.x;
  const int tid = threadIdx.x;
  if (tid < OUT_) {
    const float* w = W + tid * FC_;
    const float* x = in + m * FC_;
    float s = 0.0f;
    for (int k = 0; k < FC_; ++k) s += x[k] * w[k];
    o[tid] = s + b[tid];
  }
  __syncthreads();
  if (tid == 0) {
    float mx = o[0];
    for (int i = 1; i < OUT_; ++i) mx = fmaxf(mx, o[i]);
    float se = 0.0f;
    for (int i = 0; i < OUT_; ++i) se += __expf(o[i] - mx);
    lse = mx + logf(se);
  }
  __syncthreads();
  if (tid < OUT_) out[m * OUT_ + tid] = o[tid] - lse;
}

// ---------------- host side ----------------

extern "C" void kernel_launch(void* const* d_in, const int* in_sizes, int n_in,
                              void* d_out, int out_size, void* d_ws, size_t ws_size,
                              hipStream_t stream) {
  const float* lm_seq = (const float*)d_in[0];
  const float* W_ih0  = (const float*)d_in[1];
  const float* W_hh0  = (const float*)d_in[2];
  const float* b_ih0  = (const float*)d_in[3];
  const float* b_hh0  = (const float*)d_in[4];
  const float* W_ih1  = (const float*)d_in[5];
  const float* W_hh1  = (const float*)d_in[6];
  const float* b_ih1  = (const float*)d_in[7];
  const float* b_hh1  = (const float*)d_in[8];
  const float* W_fc1  = (const float*)d_in[9];
  const float* b_fc1  = (const float*)d_in[10];
  const float* W_fc2  = (const float*)d_in[11];
  const float* b_fc2  = (const float*)d_in[12];
  float* outp = (float*)d_out;

  // workspace carve-out (256B aligned)
  char* base = (char*)d_ws;
  size_t off = 0;
  auto carve = [&](size_t bytes) -> char* {
    char* p = base + off;
    off += (bytes + 255) & ~(size_t)255;
    return p;
  };
  __bf16* xbf    = (__bf16*)carve((size_t)B_ * T_ * D_ * 2);      // 32 MB
  __bf16* wih0b  = (__bf16*)carve((size_t)G4_ * D_ * 2);          // 4 MB
  __bf16* whh0b  = (__bf16*)carve((size_t)G4_ * H_ * 2);          // 8 MB
  __bf16* wih1b  = (__bf16*)carve((size_t)G4_ * H_ * 2);          // 8 MB
  __bf16* whh1b  = (__bf16*)carve((size_t)G4_ * H_ * 2);          // 8 MB
  float*  bsum0  = (float*) carve((size_t)G4_ * 4);
  float*  bsum1  = (float*) carve((size_t)G4_ * 4);
  __bf16* hbuf0  = (__bf16*)carve((size_t)B_ * H_ * 2);
  __bf16* hbuf1  = (__bf16*)carve((size_t)B_ * H_ * 2);
  float*  cbuf0  = (float*) carve((size_t)B_ * H_ * 4);
  float*  cbuf1  = (float*) carve((size_t)B_ * H_ * 4);
  __bf16* hs0    = (__bf16*)carve((size_t)B_ * T_ * H_ * 2);      // 64 MB
  __bf16* hs1    = (__bf16*)carve((size_t)B_ * T_ * H_ * 2);      // 64 MB
  float*  fc1out = (float*) carve((size_t)B_ * FC_ * 4);
  (void)ws_size; (void)in_sizes; (void)n_in; (void)out_size;

  __bf16* hb[2] = {hbuf0, hbuf1};
  float*  cb[2] = {cbuf0, cbuf1};

  // 1) convert inputs / weights to bf16, fold biases
  {
    size_t nx = (size_t)B_ * T_ * D_;
    k_f32_to_bf16<<<4096, 256, 0, stream>>>(lm_seq, xbf, nx);
    k_f32_to_bf16<<<2048, 256, 0, stream>>>(W_ih0, wih0b, (size_t)G4_ * D_);
    k_f32_to_bf16<<<4096, 256, 0, stream>>>(W_hh0, whh0b, (size_t)G4_ * H_);
    k_f32_to_bf16<<<4096, 256, 0, stream>>>(W_ih1, wih1b, (size_t)G4_ * H_);
    k_f32_to_bf16<<<4096, 256, 0, stream>>>(W_hh1, whh1b, (size_t)G4_ * H_);
    k_bias_sum<<<G4_ / 256, 256, 0, stream>>>(b_ih0, b_hh0, bsum0, G4_);
    k_bias_sum<<<G4_ / 256, 256, 0, stream>>>(b_ih1, b_hh1, bsum1, G4_);
  }

  // 2) layer 0: 512 sequential WMMA steps
  k_zero_state<<<(B_ * H_) / 256, 256, 0, stream>>>(hb[0], cb[0], B_ * H_);
  for (int t = 0; t < T_; ++t) {
    k_lstm_step<D_><<<H_ / 16, 128, 0, stream>>>(
        xbf, wih0b, whh0b, bsum0,
        hb[t & 1], cb[t & 1], hb[(t + 1) & 1], cb[(t + 1) & 1], hs0, t);
  }

  // 3) layer 1: 512 sequential WMMA steps on hs0
  k_zero_state<<<(B_ * H_) / 256, 256, 0, stream>>>(hb[0], cb[0], B_ * H_);
  for (int t = 0; t < T_; ++t) {
    k_lstm_step<H_><<<H_ / 16, 128, 0, stream>>>(
        hs0, wih1b, whh1b, bsum1,
        hb[t & 1], cb[t & 1], hb[(t + 1) & 1], cb[(t + 1) & 1], hs1, t);
  }

  // 4) FC head + log_softmax
  k_fc1<<<B_, 128, 0, stream>>>(hs1, W_fc1, b_fc1, fc1out);
  k_fc2_logsoftmax<<<B_, 32, 0, stream>>>(fc1out, W_fc2, b_fc2, outp);
}
